// DynamicDilatedConv_69148973465902
// MI455X (gfx1250) — matrix operand order repaired and checked by reading, not compiled, not run on hardware
//
#include <hip/hip_runtime.h>
#include <hip/hip_bf16.h>

// ---------------------------------------------------------------------------
// Deformable conv for MI455X (gfx1250, wave32, WMMA + Tensor Data Mover).
// Stage 1: NCHW -> NHWC copy of x (coalesced bilinear gathers later).
// Stage 2: offset = conv3x3(x, w_off) + b_off, stored [b][h][w][18].
// Stage 2b: w_def fp32 -> f16, tap-major [kk][o][c] (TDM-friendly 8KB tiles).
// Stage 3: bilinear sampling (fp32 math, f16 tiles) -> v_wmma_f32_16x16x32_f16
//          over K = C*K2 = 576 with f32 accumulation; per-tap weight tiles
//          DMA'd into double-buffered LDS by the TDM while VALU samples.
// ---------------------------------------------------------------------------

#define BD 8
#define CD 64
#define OD 64
#define HD 128
#define WD 128

typedef __attribute__((ext_vector_type(16))) _Float16 v16h;
typedef __attribute__((ext_vector_type(8)))  _Float16 v8h;
typedef __attribute__((ext_vector_type(4)))  _Float16 v4h;
typedef __attribute__((ext_vector_type(8)))  float    v8f;
typedef unsigned int u32x4 __attribute__((ext_vector_type(4)));
typedef int          i32x8 __attribute__((ext_vector_type(8)));
typedef int          i32x4 __attribute__((ext_vector_type(4)));

union V16U { v16h v; v8h h[2]; };

#if defined(__has_builtin)
#if __has_builtin(__builtin_amdgcn_tensor_load_to_lds) && \
    __has_builtin(__builtin_amdgcn_s_wait_tensorcnt)
#define USE_TDM 1
#endif
#endif
#ifndef USE_TDM
#define USE_TDM 0
#endif

// ------------------------- Stage 1: NCHW -> NHWC ---------------------------
__global__ __launch_bounds__(256) void k_nchw2nhwc(const float* __restrict__ x,
                                                   float* __restrict__ xn) {
    __shared__ float tile[CD * 129];        // pad to kill bank conflicts
    int bh = blockIdx.x;                    // b*H + h
    int b = bh / HD, h = bh % HD;
    for (int i = threadIdx.x; i < CD * WD; i += 256) {
        int c = i >> 7, w = i & 127;        // coalesced read along w
        tile[c * 129 + w] = x[((b * CD + c) * HD + h) * WD + w];
    }
    __syncthreads();
    for (int i = threadIdx.x; i < CD * WD; i += 256) {
        int w = i >> 6, c = i & 63;         // coalesced write along c
        xn[(bh * WD + w) * CD + c] = tile[c * 129 + w];
    }
}

// ------------------------- Stage 2: offset conv ----------------------------
__global__ __launch_bounds__(256) void k_offset_conv(const float* __restrict__ x,
                                                     const float* __restrict__ w_off,
                                                     const float* __restrict__ b_off,
                                                     float* __restrict__ off) {
    // weights re-laid as [(c*9+tap)][j], padded 18 -> 20 floats for b128 reads
    __shared__ __align__(16) float wl[576 * 20];
    for (int i = threadIdx.x; i < 10368; i += 256) {
        int j = i / 576, r = i % 576;       // w_off linear = j*576 + (c*9+tap)
        wl[r * 20 + j] = w_off[i];
    }
    __syncthreads();

    int g = blockIdx.x * 256 + threadIdx.x; // pixel id, exact cover
    int b = g / (HD * WD);
    int rem = g % (HD * WD);
    int h = rem / WD, w = rem % WD;

    float acc[18];
#pragma unroll
    for (int j = 0; j < 18; ++j) acc[j] = b_off[j];

    for (int ky = 0; ky < 3; ++ky) {
        int y = h - 1 + ky;
        if (y < 0 || y >= HD) continue;
        for (int kx = 0; kx < 3; ++kx) {
            int xx = w - 1 + kx;
            if (xx < 0 || xx >= WD) continue;
            int tap = ky * 3 + kx;
            const float* xp = x + ((size_t)(b * CD) * HD + y) * WD + xx;
            for (int c = 0; c < CD; ++c) {
                float xv = xp[(size_t)c * (HD * WD)]; // coalesced across wave
                const float* wr = &wl[(c * 9 + tap) * 20];
                float4 q0 = *(const float4*)(wr + 0);   // LDS b128 broadcast
                float4 q1 = *(const float4*)(wr + 4);
                float4 q2 = *(const float4*)(wr + 8);
                float4 q3 = *(const float4*)(wr + 12);
                float  qa = wr[16], qb = wr[17];
                acc[0]  = fmaf(xv, q0.x, acc[0]);
                acc[1]  = fmaf(xv, q0.y, acc[1]);
                acc[2]  = fmaf(xv, q0.z, acc[2]);
                acc[3]  = fmaf(xv, q0.w, acc[3]);
                acc[4]  = fmaf(xv, q1.x, acc[4]);
                acc[5]  = fmaf(xv, q1.y, acc[5]);
                acc[6]  = fmaf(xv, q1.z, acc[6]);
                acc[7]  = fmaf(xv, q1.w, acc[7]);
                acc[8]  = fmaf(xv, q2.x, acc[8]);
                acc[9]  = fmaf(xv, q2.y, acc[9]);
                acc[10] = fmaf(xv, q2.z, acc[10]);
                acc[11] = fmaf(xv, q2.w, acc[11]);
                acc[12] = fmaf(xv, q3.x, acc[12]);
                acc[13] = fmaf(xv, q3.y, acc[13]);
                acc[14] = fmaf(xv, q3.z, acc[14]);
                acc[15] = fmaf(xv, q3.w, acc[15]);
                acc[16] = fmaf(xv, qa,   acc[16]);
                acc[17] = fmaf(xv, qb,   acc[17]);
            }
        }
    }
    float* op = off + (size_t)g * 18;
#pragma unroll
    for (int j = 0; j < 18; ++j) op[j] = acc[j];
}

// ------------------ Stage 2b: w_def -> f16, tap-major ----------------------
__global__ __launch_bounds__(256) void k_prep_weights(const float* __restrict__ w_def,
                                                      _Float16* __restrict__ wh) {
    int i = blockIdx.x * 256 + threadIdx.x;   // 9*64*64 = 36864, exact cover
    int kk = i / 4096, oc = i % 4096;         // oc = o*64 + c
    wh[i] = (_Float16)w_def[oc * 9 + kk];     // wh[kk][o][c]
}

// ------- per-tap weight tile stage: TDM DMA (preferred) or LDS copy --------
__device__ __forceinline__ void stage_tap(const _Float16* __restrict__ wh, int tap,
                                          _Float16* dstLds, int t) {
    if (t >= 32) return;                      // wave 0 only
#if USE_TDM
    // D# descriptor: 1-row tile of 4096 f16 elements (8KB contiguous copy).
    // LDS aperture: generic addr low 32 bits == LDS byte offset (ISA 10.2).
    unsigned lds_addr = (unsigned)(unsigned long long)(const void*)dstLds;
    unsigned long long ga = (unsigned long long)(const void*)(wh + (size_t)tap * 4096);
    u32x4 g0 = {1u,                                   // count=1 (valid D#)
                lds_addr,                             // lds_addr [63:32]
                (unsigned)ga,                         // global_addr [95:64]
                (unsigned)((ga >> 32) & 0x01FFFFFFull) | 0x80000000u}; // +type=2
    i32x8 g1 = {0x00010000,   // data_size=1 (2B), mask=0
                0x10000000,   // tensor_dim0=4096 (low16 in bits 63:48)
                0x00010000,   // tensor_dim1=1 (bits 95:80)
                0x10000000,   // tile_dim0=4096 (bits 127:112)
                0x00000001,   // tile_dim1=1
                0x00001000,   // tensor_dim0_stride=4096
                0x10000000,   // tensor_dim1_stride=4096 (bits 223+)
                0x00000000};
    i32x4 z4 = {0, 0, 0, 0};
#if __clang_major__ >= 23
    i32x8 z8 = {0, 0, 0, 0, 0, 0, 0, 0};
    __builtin_amdgcn_tensor_load_to_lds(g0, g1, z4, z4, z8, 0);
#else
    __builtin_amdgcn_tensor_load_to_lds(g0, g1, z4, z4, 0);
#endif
#else
    const v8h* src = (const v8h*)(wh + (size_t)tap * 4096);
    v8h* dst = (v8h*)dstLds;
    for (int i = t; i < 512; i += 32) dst[i] = src[i];  // b128 copies
#endif
}

// ------------------- Stage 3: deformable conv via WMMA ---------------------
// Block: 256 threads (8 wave32), owns 64 pixels (row segment) x 64 outputs.
// M axis = output channel, N axis = pixel (coalesced NCHW stores).
__global__ __launch_bounds__(256) void k_deform_wmma(const float* __restrict__ xn,
                                                     const float* __restrict__ off,
                                                     const _Float16* __restrict__ wh,
                                                     const float* __restrict__ b_def,
                                                     float* __restrict__ out) {
    __shared__ __align__(16) _Float16 Asmp[64 * 64];      // sampled tile [p][c]
    __shared__ __align__(16) _Float16 Wbuf[2 * 64 * 64];  // double-buffered [o][c]
    __shared__ int4   cbase4[64];                          // bilinear corner bases
    __shared__ float4 cwgt4[64];                           // bilinear corner wgts

    int blk = blockIdx.x;                // B * H * (W/64) = 2048 blocks
    int b   = blk >> 8;
    int rem = blk & 255;
    int h   = rem >> 1;
    int w0  = (rem & 1) << 6;

    int t  = threadIdx.x;
    int ln = t & 31;
    int wv = t >> 5;

    // wave wv owns tiles T0=wv, T1=wv+8; same p-tile (nt), different o-tiles.
    int mt0 = wv >> 2, mt1 = (wv + 8) >> 2, nt = wv & 3;

    v8f acc0 = {}; v8f acc1 = {};

    int lane16 = ln & 15;
    int hiA = (ln >> 4) ? 8  : 0;        // A-frag K sub-offset (lanes 16..31)
    int hiB = (ln >> 4) ? 16 : 0;        // unused; B frag also uses hiA

    stage_tap(wh, 0, Wbuf, t);           // prologue: tap 0 -> buffer 0

    for (int kk = 0; kk < 9; ++kk) {
        // ---- per-pixel bilinear corner setup (reused over all channels) ----
        if (t < 64) {
            int p = t;
            int w = w0 + p;
            const float* op = off + (((size_t)(b * HD + h) * WD + w) * 18) + 2 * kk;
            float dy = op[0], dx = op[1];
            int ky = kk / 3, kx = kk % 3;
            float py = (float)(h - 1 + ky) + dy;
            float px = (float)(w - 1 + kx) + dx;
            float y0f = floorf(py), x0f = floorf(px);
            float wy = py - y0f, wx = px - x0f;
            int y0 = (int)y0f, x0 = (int)x0f;
            int bs[4]; float gw[4];
#pragma unroll
            for (int cn = 0; cn < 4; ++cn) {
                int yy = y0 + (cn >> 1);
                int xx = x0 + (cn & 1);
                float wgt = ((cn >> 1) ? wy : 1.0f - wy) * ((cn & 1) ? wx : 1.0f - wx);
                bool valid = (yy >= 0) & (yy < HD) & (xx >= 0) & (xx < WD);
                int yc = yy < 0 ? 0 : (yy > HD - 1 ? HD - 1 : yy);
                int xc = xx < 0 ? 0 : (xx > WD - 1 ? WD - 1 : xx);
                bs[cn] = ((b * HD + yc) * WD + xc) * CD;
                gw[cn] = valid ? wgt : 0.0f;
            }
            cbase4[p] = make_int4(bs[0], bs[1], bs[2], bs[3]);
            cwgt4[p]  = make_float4(gw[0], gw[1], gw[2], gw[3]);
        }
        __syncthreads();   // corners visible; all waves past WMMA(kk-1)

        // DMA next tap's weights while we do this tap's sampling
        if (kk < 8) stage_tap(wh, kk + 1, Wbuf + (((kk + 1) & 1) << 12), t);

        // ---- bilinear sampling: float4 gathers, packed f16 b64 stores ----
        {
            int c4   = (t & 15) * 4;
            int psub = t >> 4;               // 0..15, half-wave per pixel
            for (int pg = 0; pg < 4; ++pg) {
                int p = pg * 16 + psub;
                int4   cb = cbase4[p];       // one b128 broadcast
                float4 cw = cwgt4[p];        // one b128 broadcast
                float4 s0 = *(const float4*)&xn[cb.x + c4];
                float4 s1 = *(const float4*)&xn[cb.y + c4];
                float4 s2 = *(const float4*)&xn[cb.z + c4];
                float4 s3 = *(const float4*)&xn[cb.w + c4];
                float r0 = cw.x * s0.x + cw.y * s1.x + cw.z * s2.x + cw.w * s3.x;
                float r1 = cw.x * s0.y + cw.y * s1.y + cw.z * s2.y + cw.w * s3.y;
                float r2 = cw.x * s0.z + cw.y * s1.z + cw.z * s2.z + cw.w * s3.z;
                float r3 = cw.x * s0.w + cw.y * s1.w + cw.z * s2.w + cw.w * s3.w;
                v4h hv;
                hv[0] = (_Float16)r0; hv[1] = (_Float16)r1;
                hv[2] = (_Float16)r2; hv[3] = (_Float16)r3;
                *(v4h*)&Asmp[p * 64 + c4] = hv;   // ds_store_b64
            }
        }
#if USE_TDM
        if (t < 32) {                        // tap kk's DMA must have landed
            if (kk < 8) __builtin_amdgcn_s_wait_tensorcnt(1);
            else        __builtin_amdgcn_s_wait_tensorcnt(0);
        }
#endif
        __syncthreads();   // Asmp + weight tile visible to all

        // ---- matrix stage: D[o,p] += A[o,k] * B[k,p], K = 64 (2 chunks) ----
        const _Float16* Wcur = Wbuf + ((kk & 1) << 12);
#pragma unroll
        for (int cb = 0; cb < 64; cb += 32) {
            V16U a0, a1, bb;
            int colp = nt * 16 + lane16;     // B fragment shared by both tiles
            bb.h[0] = *(const v8h*)&Asmp[colp * 64 + cb + hiA];
            bb.h[1] = *(const v8h*)&Asmp[colp * 64 + cb + hiA + 16];
            int rowo0 = mt0 * 16 + lane16;
            a0.h[0] = *(const v8h*)&Wcur[rowo0 * 64 + cb + hiA];
            a0.h[1] = *(const v8h*)&Wcur[rowo0 * 64 + cb + hiA + 16];
            acc0 = __builtin_amdgcn_wmma_f32_16x16x32_f16(
                false, a0.v, false, bb.v, (short)0, acc0, false, false);
            int rowo1 = mt1 * 16 + lane16;
            a1.h[0] = *(const v8h*)&Wcur[rowo1 * 64 + cb + hiA];
            a1.h[1] = *(const v8h*)&Wcur[rowo1 * 64 + cb + hiA + 16];
            acc1 = __builtin_amdgcn_wmma_f32_16x16x32_f16(
                false, a1.v, false, bb.v, (short)0, acc1, false, false);
        }
    }

    // ---- bias + NCHW store; lanes map to consecutive w -> coalesced ----
    int mhi = (ln >> 4) ? 8 : 0;
    int wloc = w0 + nt * 16 + lane16;
#pragma unroll
    for (int r = 0; r < 8; ++r) {
        int o = mt0 * 16 + r + mhi;
        out[((size_t)(b * OD + o) * HD + h) * WD + wloc] = acc0[r] + b_def[o];
    }
#pragma unroll
    for (int r = 0; r < 8; ++r) {
        int o = mt1 * 16 + r + mhi;
        out[((size_t)(b * OD + o) * HD + h) * WD + wloc] = acc1[r] + b_def[o];
    }
}

// ---------------------------------------------------------------------------
extern "C" void kernel_launch(void* const* d_in, const int* in_sizes, int n_in,
                              void* d_out, int out_size, void* d_ws, size_t ws_size,
                              hipStream_t stream) {
    const float* x     = (const float*)d_in[0];  // (8,64,128,128)
    const float* w_off = (const float*)d_in[1];  // (18,64,3,3)
    const float* b_off = (const float*)d_in[2];  // (18,)
    const float* w_def = (const float*)d_in[3];  // (64,64,3,3)
    const float* b_def = (const float*)d_in[4];  // (64,)
    float* out = (float*)d_out;                  // (8,64,128,128)

    // workspace: NHWC x (33.5MB) + offsets (9.4MB) + f16 weights (72KB)
    float*    xn  = (float*)d_ws;
    float*    off = xn + (size_t)BD * HD * WD * CD;
    _Float16* wh  = (_Float16*)(off + (size_t)BD * HD * WD * 18);

    k_nchw2nhwc  <<<BD * HD, 256, 0, stream>>>(x, xn);
    k_offset_conv<<<(BD * HD * WD) / 256, 256, 0, stream>>>(x, w_off, b_off, off);
    k_prep_weights<<<(9 * OD * CD) / 256, 256, 0, stream>>>(w_def, wh);
    k_deform_wmma<<<BD * HD * (WD / 64), 256, 0, stream>>>(xn, off, wh, b_def, out);
}
